// DCTBlur_36163624632509
// MI455X (gfx1250) — matrix operand order repaired and checked by reading, not compiled, not run on hardware
//
#include <hip/hip_runtime.h>

// DCT blur:  out = M_s X M_s^T  with  M_s = D diag(e_s) D  (exp mask is separable).
// k_build_d:  D (256x256 f32) into workspace.
// k_build_m:  M_s table, 32 x 256x256 f16, into workspace (4 MB).
// k_dctblur:  one workgroup (8 wave32) per image; fused GEMM-GEMM through 264 KB LDS,
//             f16 WMMA (V_WMMA_F32_16X16X32_F16), f32 accumulate, transposed stores.

#define NN 256
#define RS 264          // padded LDS row stride in halfs (528 B -> conflict-free b128)
#define PI_F 3.14159265358979323846f

typedef __attribute__((ext_vector_type(16))) _Float16    v16h;
typedef __attribute__((ext_vector_type(2)))  _Float16    v2h;
typedef __attribute__((ext_vector_type(8)))  float       v8f;
typedef __attribute__((ext_vector_type(4)))  float       v4f;
typedef __attribute__((ext_vector_type(4)))  unsigned int v4u;
typedef __attribute__((ext_vector_type(2)))  unsigned int v2u;

union HU { v4u q[2]; v16h h; };

__device__ __forceinline__ unsigned int pkh2(float a, float b) {
    v2h p; p.x = (_Float16)a; p.y = (_Float16)b;
    return __builtin_bit_cast(unsigned int, p);
}

// ---------------- setup kernel 1: orthonormal DCT-II matrix ----------------
__global__ __launch_bounds__(256) void k_build_d(float* __restrict__ D) {
    int k = blockIdx.x, n = threadIdx.x;
    float v = sqrtf(2.0f / 256.0f) * cosf(PI_F * (n + 0.5f) * (float)k / 256.0f);
    if (k == 0) v *= 0.70710678118654752f;
    D[k * NN + n] = v;
}

// ---------------- setup kernel 2: M_s = D diag(e_s) D  (f16 table) ----------------
__global__ __launch_bounds__(256) void k_build_m(const float* __restrict__ D,
                                                 const float* __restrict__ sigmas,
                                                 _Float16* __restrict__ Mtab) {
    __shared__ float e[NN];
    const int s   = blockIdx.z;
    const int tid = threadIdx.y * 16 + threadIdx.x;
    {
        float sig = sigmas[s];
        float t   = 0.5f * sig * sig;
        float f   = PI_F * (float)tid / 256.0f;
        e[tid] = expf(-f * f * t);
    }
    __syncthreads();
    const int a = blockIdx.y * 16 + threadIdx.y;
    const int b = blockIdx.x * 16 + threadIdx.x;
    float acc = 0.0f;
    for (int k = 0; k < NN; ++k)
        acc = fmaf(D[a * NN + k] * e[k], D[k * NN + b], acc);
    Mtab[(size_t)s * (NN * NN) + a * NN + b] = (_Float16)acc;
}

// ---------------- main kernel: out = M X M^T, fused through LDS ----------------
__global__ __launch_bounds__(256) void k_dctblur(const float*    __restrict__ x,
                                                 const _Float16* __restrict__ Mtab,
                                                 const int*      __restrict__ steps,
                                                 float*          __restrict__ out) {
    __shared__ __align__(16) unsigned short Xb[NN * RS];   // image, f16, row-major
    __shared__ __align__(16) unsigned short TT[NN * RS];   // (M X)^T, f16, row-major

    const int img   = blockIdx.x;          // b*3 + c
    const int batch = img / 3;
    const _Float16* Ms = Mtab + (size_t)steps[batch] * (NN * NN);
    const float*    xi = x   + (size_t)img * (NN * NN);
    float*          oi = out + (size_t)img * (NN * NN);

    const int t    = threadIdx.x;
    const int lane = t & 31;
    const int wave = t >> 5;

    // ---- phase 0: f32 -> f16 into LDS (coalesced float4 loads) ----
    for (int i = 0; i < 64; ++i) {
        int flat = (i * 256 + t) * 4;
        int row  = flat >> 8, col = flat & 255;
        v4f v = *(const v4f*)(xi + flat);
        v2u p; p.x = pkh2(v.x, v.y); p.y = pkh2(v.z, v.w);
        *(v2u*)&Xb[row * RS + col] = p;
    }
    __syncthreads();

    const v4u* Mq    = (const v4u*)Ms;        // 16-B units, row stride = 32 units
    const int  arow  = lane & 15;             // A-operand row within stripe
    const int  ahalf = (lane < 16) ? 0 : 1;   // K-half select (0..7 vs 8..15)
    const int  ncol  = lane & 15;             // C/D-tile column (lane)
    const int  moff  = (lane < 16) ? 0 : 8;   // C/D-tile row offset

    // ---- sweep 1: TT = (M @ X)^T ----
    for (int rep = 0; rep < 2; ++rep) {
        const int I0 = (wave + 8 * rep) * 16;      // output row stripe
        HU A[8];                                    // 16x256 f16 M stripe in 64 VGPRs
#pragma unroll
        for (int kt = 0; kt < 8; ++kt) {
            int qb = (I0 + arow) * 32 + kt * 4 + ahalf;
            A[kt].q[0] = Mq[qb];
            A[kt].q[1] = Mq[qb + 2];
        }
        for (int jb = 0; jb < 16; ++jb) {
            const int J0 = jb * 16;
            v8f acc = {};
#pragma unroll
            for (int kt = 0; kt < 8; ++kt) {
                HU B;
                int kr = kt * 32 + ahalf * 8;       // B row = K index
#pragma unroll
                for (int s = 0; s < 8; ++s) {
                    B.h[s]     = __builtin_bit_cast(_Float16, Xb[(kr + s) * RS + J0 + ncol]);
                    B.h[s + 8] = __builtin_bit_cast(_Float16, Xb[(kr + 16 + s) * RS + J0 + ncol]);
                }
                acc = __builtin_amdgcn_wmma_f32_16x16x32_f16(
                        false, A[kt].h, false, B.h, (short)0, acc, false, false);
            }
            // transposed f16 store: lane's 8 elems are one contiguous LDS run
            v4u pk;
            pk.x = pkh2(acc[0], acc[1]); pk.y = pkh2(acc[2], acc[3]);
            pk.z = pkh2(acc[4], acc[5]); pk.w = pkh2(acc[6], acc[7]);
            *(v4u*)&TT[(J0 + ncol) * RS + I0 + moff] = pk;
        }
    }
    __syncthreads();

    // ---- sweep 2: out = (M @ TT)^T  (= T M^T), f32 straight to HBM ----
    for (int rep = 0; rep < 2; ++rep) {
        const int A0 = (wave + 8 * rep) * 16;
        HU A[8];
#pragma unroll
        for (int kt = 0; kt < 8; ++kt) {
            int qb = (A0 + arow) * 32 + kt * 4 + ahalf;
            A[kt].q[0] = Mq[qb];
            A[kt].q[1] = Mq[qb + 2];
        }
        for (int jb = 0; jb < 16; ++jb) {
            const int J0 = jb * 16;
            v8f acc = {};
#pragma unroll
            for (int kt = 0; kt < 8; ++kt) {
                HU B;
                int kr = kt * 32 + ahalf * 8;
#pragma unroll
                for (int s = 0; s < 8; ++s) {
                    B.h[s]     = __builtin_bit_cast(_Float16, TT[(kr + s) * RS + J0 + ncol]);
                    B.h[s + 8] = __builtin_bit_cast(_Float16, TT[(kr + 16 + s) * RS + J0 + ncol]);
                }
                acc = __builtin_amdgcn_wmma_f32_16x16x32_f16(
                        false, A[kt].h, false, B.h, (short)0, acc, false, false);
            }
            // transposed f32 store: out[J0+ncol, A0+moff .. +3 / +4 .. +7]
            float* p = oi + (J0 + ncol) * NN + A0 + moff;
            v4f lo = {acc[0], acc[1], acc[2], acc[3]};
            v4f hi = {acc[4], acc[5], acc[6], acc[7]};
            *(v4f*)p       = lo;
            *((v4f*)p + 1) = hi;
        }
    }
}

extern "C" void kernel_launch(void* const* d_in, const int* in_sizes, int n_in,
                              void* d_out, int out_size, void* d_ws, size_t ws_size,
                              hipStream_t stream) {
    (void)in_sizes; (void)n_in; (void)out_size; (void)ws_size;
    const float* x      = (const float*)d_in[0];
    const float* sigmas = (const float*)d_in[1];
    const int*   steps  = (const int*)d_in[2];
    float*       out    = (float*)d_out;

    float*    D    = (float*)d_ws;                                   // 256 KB
    _Float16* Mtab = (_Float16*)((char*)d_ws + NN * NN * sizeof(float)); // 4 MB

    k_build_d<<<NN, NN, 0, stream>>>(D);
    k_build_m<<<dim3(16, 16, 32), dim3(16, 16), 0, stream>>>(D, sigmas, Mtab);
    k_dctblur<<<128 * 3, 256, 0, stream>>>(x, Mtab, steps, out);
}
